// ResidualCGConvBlock_52862457480029
// MI455X (gfx1250) — matrix operand order repaired and checked by reading, not compiled, not run on hardware
//
#include <hip/hip_runtime.h>
#include <hip/hip_bf16.h>

// ---------------------------------------------------------------------------
// ResidualCGConvBlock on MI455X (gfx1250, wave32, WMMA bf16 16x16x32)
//
// Math:  msg = sigmoid(z@Wf^T+bf) * softplus(z@Ws^T+bs),
//        z = [x_dst | x_src | e_attr]  (128+128+16)
// Linear decomposition: z@W^T = Pdst[dst] + Psrc[src] + e_attr@We^T
//   -> 4 node GEMMs [20000,128]x[128,128] (WMMA bf16) into L2-resident tables
//   -> per-edge 16xK=32 WMMA for the edge-attr part
//   -> gather + activations + global_atomic_add_f32 segment sum
// ---------------------------------------------------------------------------

#define NNODES 20000
#define FEAT   128
#define EDIM   16
#define NEDGE  600000
#define KPAD   288          // 272 padded to 9*32
#define EPSV   1e-5f

typedef __bf16 bf16_t;
typedef __attribute__((ext_vector_type(16))) __bf16 v16bf;
typedef __attribute__((ext_vector_type(8)))  __bf16 v8bf;
typedef __attribute__((ext_vector_type(8)))  float  v8f;

// ---- WMMA fragment loaders (layouts per CDNA5 ISA 7.12.2) -------------------
// A (16x32 bf16): lane row = lane&15; lane<16 holds K {ks+0..7, ks+16..23},
//                 lane>=16 holds K {ks+8..15, ks+24..31}
__device__ __forceinline__ v16bf load_a_frag(const bf16_t* base, int ldk,
                                             int row, int kstep, int lane) {
    const int khalf = (lane >> 4) * 8;
    const bf16_t* p = base + row * ldk + kstep + khalf;
    v8bf c0 = *(const v8bf*)(p);
    v8bf c1 = *(const v8bf*)(p + 16);
    v16bf a;
#pragma unroll
    for (int i = 0; i < 8; ++i) { a[i] = c0[i]; a[i + 8] = c1[i]; }
    return a;
}

// B (32x16 bf16): lane column = lane&15; lane<16 holds K {ks+0..7, ks+16..23},
// lane>=16 holds K {ks+8..15, ks+24..31}; weights row-major [out, KPAD].
__device__ __forceinline__ v16bf load_b_frag(const bf16_t* W, int ldk,
                                             int col0, int kstep, int lane) {
    const int col = col0 + (lane & 15);
    const int k0  = kstep + ((lane >> 4) << 3);
    const bf16_t* p = W + col * ldk + k0;
    v8bf c0 = *(const v8bf*)(p);
    v8bf c1 = *(const v8bf*)(p + 16);
    v16bf b;
#pragma unroll
    for (int i = 0; i < 8; ++i) { b[i] = c0[i]; b[i + 8] = c1[i]; }
    return b;
}

__device__ __forceinline__ v8f wmma_bf16(v16bf a, v16bf b, v8f c) {
    return __builtin_amdgcn_wmma_f32_16x16x32_bf16(
        /*neg_a=*/false, a, /*neg_b=*/false, b,
        /*c_mod=*/(short)0, c, /*reuse_a=*/false, /*reuse_b=*/false);
}

// ---- Kernel 1: x f32 -> bf16, zero agg -------------------------------------
__global__ void k_prep_x(const float* __restrict__ x, bf16_t* __restrict__ xbf,
                         float* __restrict__ agg, int total) {
    int i = blockIdx.x * blockDim.x + threadIdx.x;
    if (i < total) { xbf[i] = (bf16_t)x[i]; agg[i] = 0.0f; }
}

// ---- Kernel 2: weights -> bf16 [128, KPAD] with zero pad, zero stats -------
__global__ void k_prep_w(const float* __restrict__ Wf, const float* __restrict__ Ws,
                         bf16_t* __restrict__ Wfb, bf16_t* __restrict__ Wsb,
                         float* __restrict__ stats) {
    int i = blockIdx.x * blockDim.x + threadIdx.x;
    if (i < 256) stats[i] = 0.0f;
    if (i < FEAT * KPAD) {
        int o = i / KPAD, k = i - o * KPAD;
        float vf = (k < 272) ? Wf[o * 272 + k] : 0.0f;
        float vs = (k < 272) ? Ws[o * 272 + k] : 0.0f;
        Wfb[i] = (bf16_t)vf;
        Wsb[i] = (bf16_t)vs;
    }
}

// ---- Kernel 3: node GEMMs. Block = 4 waves; wave w computes table w --------
// tables: 0 = x@Wf[:,0:128]^T (dst,f), 1 = x@Wf[:,128:256]^T (src,f),
//         2 = x@Ws[:,0:128]^T (dst,s), 3 = x@Ws[:,128:256]^T (src,s)
__global__ void __launch_bounds__(128)
k_node_gemm(const bf16_t* __restrict__ xbf,
            const bf16_t* __restrict__ Wfb, const bf16_t* __restrict__ Wsb,
            float* __restrict__ Pfd, float* __restrict__ Pfs,
            float* __restrict__ Psd, float* __restrict__ Pss) {
    const int wave = threadIdx.x >> 5;
    const int lane = threadIdx.x & 31;
    const int row0 = blockIdx.x * 16;               // 20000/16 = 1250 exact

    const bf16_t* W = (wave < 2) ? Wfb : Wsb;
    const int koff  = (wave & 1) ? 128 : 0;
    float* P = (wave == 0) ? Pfd : (wave == 1) ? Pfs : (wave == 2) ? Psd : Pss;

    // A fragments: 16 node rows x 128 features, 4 K-steps of 32 (kept in regs)
    const int arow = row0 + (lane & 15);
    v16bf afr[4];
#pragma unroll
    for (int ks = 0; ks < 4; ++ks)
        afr[ks] = load_a_frag(xbf, FEAT, arow, ks * 32, lane);

    const int c    = lane & 15;
    const int mb   = (lane >> 4) * 8;
#pragma unroll
    for (int ct = 0; ct < 8; ++ct) {
        v8f acc = {};
#pragma unroll
        for (int ks = 0; ks < 4; ++ks) {
            v16bf b = load_b_frag(W, KPAD, ct * 16, koff + ks * 32, lane);
            acc = wmma_bf16(afr[ks], b, acc);
        }
        const int col = ct * 16 + c;
#pragma unroll
        for (int v = 0; v < 8; ++v)                  // D: lane=col, vgpr=row
            P[(row0 + mb + v) * FEAT + col] = acc[v];
    }
}

// ---- Kernel 4: edge kernel. Block = 8 waves; wave handles 16 edges ---------
__global__ void __launch_bounds__(256)
k_edges(const int* __restrict__ eidx,          // [2, E] (row0=src, row1=dst)
        const float* __restrict__ eattr,       // [E, 16] f32
        const bf16_t* __restrict__ Wfb, const bf16_t* __restrict__ Wsb,
        const float* __restrict__ Pfd, const float* __restrict__ Pfs,
        const float* __restrict__ Psd, const float* __restrict__ Pss,
        const float* __restrict__ bf_, const float* __restrict__ bs_,
        float* __restrict__ agg, int nE) {
    __shared__ int s_idx[8][32];                 // per wave: [0:16)=dst, [16:32)=src
    const int wave = threadIdx.x >> 5;
    const int lane = threadIdx.x & 31;
    const int tile = blockIdx.x * 8 + wave;
    const int e0   = tile * 16;
    if (e0 >= nE) return;                        // uniform per wave, no syncthreads

    // stash this tile's indices in LDS (in-order DS within a wave)
    int idx = (lane < 16) ? eidx[nE + e0 + lane]          // dst
                          : eidx[e0 + (lane - 16)];       // src
    s_idx[wave][lane] = idx;

    // A fragment: edge_attr tile [16 edges, K=32] with K>=16 zero-padded.
    // lane<16 -> real K 0..7 (chunk0), pad K16..23; lane>=16 -> K8..15, pad.
    {
        const int r  = lane & 15;
        const int f0 = (lane >> 4) * 8;
        v8f av = *(const v8f*)(eattr + (e0 + r) * EDIM + f0);
        v16bf a;
#pragma unroll
        for (int i = 0; i < 8; ++i) { a[i] = (bf16_t)av[i]; a[i + 8] = (bf16_t)0.0f; }

        const int c  = lane & 15;
        const int mb = (lane >> 4) * 8;
#pragma unroll
        for (int ct = 0; ct < 8; ++ct) {
            v8f accf = {}, accs = {};
            v16bf bf_frag = load_b_frag(Wfb, KPAD, ct * 16, 256, lane);
            v16bf bs_frag = load_b_frag(Wsb, KPAD, ct * 16, 256, lane);
            accf = wmma_bf16(a, bf_frag, accf);
            accs = wmma_bf16(a, bs_frag, accs);

            const int col = ct * 16 + c;
            const float bfv = bf_[col];
            const float bsv = bs_[col];
#pragma unroll
            for (int v = 0; v < 8; ++v) {
                const int m = mb + v;            // edge within tile
                const int d = s_idx[wave][m];
                const int s = s_idx[wave][16 + m];
                float fl = accf[v] + Pfd[d * FEAT + col] + Pfs[s * FEAT + col] + bfv;
                float sl = accs[v] + Psd[d * FEAT + col] + Pss[s * FEAT + col] + bsv;
                float gate = __frcp_rn(1.0f + __expf(-fl));
                float sp   = fmaxf(sl, 0.0f) + __logf(1.0f + __expf(-fabsf(sl)));
                unsafeAtomicAdd(&agg[d * FEAT + col], gate * sp);
            }
        }
    }
}

// ---- Kernel 5: BatchNorm batch statistics over nodes -----------------------
__global__ void __launch_bounds__(128)
k_bn_stats(const float* __restrict__ agg, float* __restrict__ stats, int n) {
    const int f = threadIdx.x;                   // feature
    float s = 0.0f, ss = 0.0f;
    for (int r = blockIdx.x; r < n; r += gridDim.x) {
        float v = agg[r * FEAT + f];
        s += v; ss += v * v;
    }
    unsafeAtomicAdd(&stats[f], s);
    unsafeAtomicAdd(&stats[FEAT + f], ss);
}

// ---- Kernel 6: BN -> residual -> LayerNorm -> ReLU -> residual -------------
__global__ void __launch_bounds__(128)
k_finalize(const float* __restrict__ agg, const float* __restrict__ x,
           const float* __restrict__ stats,
           const float* __restrict__ bng, const float* __restrict__ bnb,
           const float* __restrict__ lng, const float* __restrict__ lnb,
           float* __restrict__ out, float invN) {
    __shared__ float sb[FEAT];
    const int node = blockIdx.x;
    const int f    = threadIdx.x;

    const float mu  = stats[f] * invN;
    const float var = stats[FEAT + f] * invN - mu * mu;
    const float xv  = x[node * FEAT + f];
    float a = (agg[node * FEAT + f] - mu) * rsqrtf(var + EPSV) * bng[f] + bnb[f];
    float h = a + xv;

    // LayerNorm over 128 features
    sb[f] = h; __syncthreads();
    for (int off = 64; off > 0; off >>= 1) {
        if (f < off) sb[f] += sb[f + off];
        __syncthreads();
    }
    const float mean = sb[0] * (1.0f / FEAT);
    __syncthreads();
    const float dv = h - mean;
    sb[f] = dv * dv; __syncthreads();
    for (int off = 64; off > 0; off >>= 1) {
        if (f < off) sb[f] += sb[f + off];
        __syncthreads();
    }
    const float lvar = sb[0] * (1.0f / FEAT);
    float ln = dv * rsqrtf(lvar + EPSV) * lng[f] + lnb[f];
    out[node * FEAT + f] = fmaxf(ln, 0.0f) + xv;
}

// ---------------------------------------------------------------------------
static inline size_t align256(size_t v) { return (v + 255) & ~size_t(255); }

extern "C" void kernel_launch(void* const* d_in, const int* in_sizes, int n_in,
                              void* d_out, int out_size, void* d_ws, size_t ws_size,
                              hipStream_t stream) {
    const float* x     = (const float*)d_in[0];
    const int*   eidx  = (const int*)d_in[1];
    const float* eattr = (const float*)d_in[2];
    const float* Wf    = (const float*)d_in[3];
    const float* bfv   = (const float*)d_in[4];
    const float* Ws    = (const float*)d_in[5];
    const float* bsv   = (const float*)d_in[6];
    const float* bng   = (const float*)d_in[7];
    const float* bnb   = (const float*)d_in[8];
    const float* lng   = (const float*)d_in[9];
    const float* lnb   = (const float*)d_in[10];
    float* out = (float*)d_out;

    // workspace carve-out (~57 MB)
    char* w = (char*)d_ws;
    bf16_t* xbf = (bf16_t*)w;  w += align256(sizeof(bf16_t) * NNODES * FEAT);
    bf16_t* Wfb = (bf16_t*)w;  w += align256(sizeof(bf16_t) * FEAT * KPAD);
    bf16_t* Wsb = (bf16_t*)w;  w += align256(sizeof(bf16_t) * FEAT * KPAD);
    float* Pfd  = (float*)w;   w += align256(sizeof(float) * NNODES * FEAT);
    float* Pfs  = (float*)w;   w += align256(sizeof(float) * NNODES * FEAT);
    float* Psd  = (float*)w;   w += align256(sizeof(float) * NNODES * FEAT);
    float* Pss  = (float*)w;   w += align256(sizeof(float) * NNODES * FEAT);
    float* agg  = (float*)w;   w += align256(sizeof(float) * NNODES * FEAT);
    float* stats = (float*)w;  w += align256(sizeof(float) * 256);

    const int total = NNODES * FEAT;
    k_prep_x<<<(total + 255) / 256, 256, 0, stream>>>(x, xbf, agg, total);
    k_prep_w<<<(FEAT * KPAD + 255) / 256, 256, 0, stream>>>(Wf, Ws, Wfb, Wsb, stats);
    k_node_gemm<<<NNODES / 16, 128, 0, stream>>>(xbf, Wfb, Wsb, Pfd, Pfs, Psd, Pss);
    k_edges<<<(NEDGE / 16 + 7) / 8, 256, 0, stream>>>(eidx, eattr, Wfb, Wsb,
                                                      Pfd, Pfs, Psd, Pss,
                                                      bfv, bsv, agg, NEDGE);
    k_bn_stats<<<256, 128, 0, stream>>>(agg, stats, NNODES);
    k_finalize<<<NNODES, 128, 0, stream>>>(agg, x, stats, bng, bnb, lng, lnb,
                                           out, 1.0f / (float)NNODES);
}
